// EdgeBlock_19877108646538
// MI455X (gfx1250) — compile-verified, matrix operand order checked by reading, not run on hardware
//
#include <hip/hip_runtime.h>
#include <stdint.h>

// ---------------------------------------------------------------------------
// EdgeBlock (graph-net edge update) for gfx1250:
//   collected = [edges | nodes[recv] | nodes[send] | globals[g]]  (E x 224, f32)
//   out = relu(collected @ W1 + b1) @ W2 + b2                     (E x 64,  f32)
// Both GEMMs run on v_wmma_f32_16x16x32_bf16 (bf16 in, f32 accumulate).
// f32->bf16 uses native casts so the backend emits hardware converts
// instead of manual shift/round VALU sequences.
// ---------------------------------------------------------------------------

typedef __attribute__((ext_vector_type(16))) __bf16        v16bf;
typedef __attribute__((ext_vector_type(2)))  __bf16        v2bf;
typedef __attribute__((ext_vector_type(8)))  float         v8f;
typedef __attribute__((ext_vector_type(4)))  unsigned int  v4u;
typedef __attribute__((ext_vector_type(4)))  float         v4f;

#define D_NODE   64
#define D_EDGE   64
#define D_GLOB   32
#define D_IN     224   // 64 + 64 + 64 + 32
#define D_HID    256
#define D_OUT    64

#define TILE_M   64          // edges per block
#define SA_STRIDE 232        // D_IN  + 8 pad: 116 dwords, coprime w/ 64 banks
#define SH_STRIDE 264        // D_HID + 8 pad: 132 dwords, conflict-free

__device__ __forceinline__ unsigned short f2bf(float f) {
    __bf16 h = (__bf16)f;                      // native cvt, RNE
    return __builtin_bit_cast(unsigned short, h);
}
__device__ __forceinline__ unsigned int pack2(float a, float b) {
    v2bf p;
    p[0] = (__bf16)a;
    p[1] = (__bf16)b;
    return __builtin_bit_cast(unsigned int, p); // packed cvt where available
}

union Frag {
    v16bf v;      // WMMA 16-bit operand: 16 bf16 per lane (8 VGPRs)
    v4u   u[2];   // two 16-byte chunks (b128 loads)
};

// --- weight prep: f32 [K][N] -> bf16 transposed [N][K] ----------------------
__global__ __launch_bounds__(256) void prep_weights(
        const float* __restrict__ W1, const float* __restrict__ W2,
        unsigned short* __restrict__ w1t, unsigned short* __restrict__ w2t) {
    const int total1 = D_HID * D_IN;    // 57344
    const int total2 = D_OUT * D_HID;   // 16384
    int idx = blockIdx.x * 256 + threadIdx.x;
    if (idx < total1) {
        int n = idx / D_IN, k = idx % D_IN;
        w1t[idx] = f2bf(W1[k * D_HID + n]);
    } else if (idx < total1 + total2) {
        int i = idx - total1;
        int n = i / D_HID, k = i % D_HID;
        w2t[i] = f2bf(W2[k * D_OUT + n]);
    }
}

// --- main fused gather + 2-layer MLP kernel --------------------------------
__global__ __launch_bounds__(256) void edge_mlp_kernel(
        const float* __restrict__ edges,
        const float* __restrict__ nodes,
        const float* __restrict__ globals_,
        const int*   __restrict__ receivers,
        const int*   __restrict__ senders,
        const int*   __restrict__ egi,
        const unsigned short* __restrict__ w1t,   // bf16 [256][224]
        const float* __restrict__ b1,
        const unsigned short* __restrict__ w2t,   // bf16 [64][256]
        const float* __restrict__ b2,
        float* __restrict__ out,
        int n_edges) {
    __shared__ __align__(16) unsigned short sA[TILE_M * SA_STRIDE]; // collected bf16
    __shared__ __align__(16) unsigned short sH[TILE_M * SH_STRIDE]; // hidden bf16

    const int t  = threadIdx.x;
    const int e0 = blockIdx.x * TILE_M;

    // ---------------- gather + f32->bf16 convert into LDS -----------------
    {
        const int row = t >> 2;           // edge row within tile (0..63)
        const int q   = t & 3;            // quarter of the feature vector
        int e = e0 + row;
        if (e >= n_edges) e = n_edges - 1;            // clamp (uniform per row)
        const int r = receivers[e];
        const int s = senders[e];
        const int g = egi[e];

        // edge features -> cols [0,64)
        {
            const v4f* src = (const v4f*)(edges + (size_t)e * D_EDGE + q * 16);
            unsigned int* dst = (unsigned int*)&sA[row * SA_STRIDE + q * 16];
#pragma unroll
            for (int i = 0; i < 4; ++i) {
                v4f f = src[i];
                dst[2 * i]     = pack2(f[0], f[1]);
                dst[2 * i + 1] = pack2(f[2], f[3]);
            }
        }
        // receiver node features -> cols [64,128)
        {
            const v4f* src = (const v4f*)(nodes + (size_t)r * D_NODE + q * 16);
            unsigned int* dst = (unsigned int*)&sA[row * SA_STRIDE + 64 + q * 16];
#pragma unroll
            for (int i = 0; i < 4; ++i) {
                v4f f = src[i];
                dst[2 * i]     = pack2(f[0], f[1]);
                dst[2 * i + 1] = pack2(f[2], f[3]);
            }
        }
        // sender node features -> cols [128,192)
        {
            const v4f* src = (const v4f*)(nodes + (size_t)s * D_NODE + q * 16);
            unsigned int* dst = (unsigned int*)&sA[row * SA_STRIDE + 128 + q * 16];
#pragma unroll
            for (int i = 0; i < 4; ++i) {
                v4f f = src[i];
                dst[2 * i]     = pack2(f[0], f[1]);
                dst[2 * i + 1] = pack2(f[2], f[3]);
            }
        }
        // per-graph globals -> cols [192,224)
        {
            const v4f* src = (const v4f*)(globals_ + (size_t)g * D_GLOB + q * 8);
            unsigned int* dst = (unsigned int*)&sA[row * SA_STRIDE + 192 + q * 8];
#pragma unroll
            for (int i = 0; i < 2; ++i) {
                v4f f = src[i];
                dst[2 * i]     = pack2(f[0], f[1]);
                dst[2 * i + 1] = pack2(f[2], f[3]);
            }
        }
    }
    __syncthreads();

    const int wave = t >> 5;
    const int lane = t & 31;
    const int lrow = lane & 15;        // A-row / B-col / C-col within tile
    const int lhi  = lane >> 4;        // which K half this lane holds
    const int koff = lhi * 8;          // per-lane K sub-offset (bf16 elems)

    // ---------------- stage 1: h = relu(collected @ W1 + b1) ---------------
    // wave owns hidden N-tiles {2*wave, 2*wave+1} x all 4 M-subtiles.
    const int nt0 = wave * 2;
    v8f acc[4][2];
#pragma unroll
    for (int m = 0; m < 4; ++m)
#pragma unroll
        for (int j = 0; j < 2; ++j) acc[m][j] = {};

#pragma unroll
    for (int ks = 0; ks < D_IN / 32; ++ks) {           // 7 K-steps of 32
        const int kc = ks * 32;
        Frag b[2];
#pragma unroll
        for (int j = 0; j < 2; ++j) {                  // W1T B fragments
            const unsigned short* bp =
                w1t + (size_t)((nt0 + j) * 16 + lrow) * D_IN + kc + koff;
            b[j].u[0] = *(const v4u*)bp;               // K {kc+koff .. +7}
            b[j].u[1] = *(const v4u*)(bp + 16);        // K {kc+16+koff .. +7}
        }
#pragma unroll
        for (int m = 0; m < 4; ++m) {
            Frag a;
            const unsigned short* ap =
                &sA[(m * 16 + lrow) * SA_STRIDE + kc + koff];
            a.u[0] = *(const v4u*)ap;
            a.u[1] = *(const v4u*)(ap + 16);
#pragma unroll
            for (int j = 0; j < 2; ++j)
                acc[m][j] = __builtin_amdgcn_wmma_f32_16x16x32_bf16(
                    false, a.v, false, b[j].v, (short)0, acc[m][j], false, false);
        }
    }

    // bias + relu, convert to bf16 h-tile in LDS
    {
        float bias1[2];
#pragma unroll
        for (int j = 0; j < 2; ++j) bias1[j] = b1[(nt0 + j) * 16 + lrow];
#pragma unroll
        for (int m = 0; m < 4; ++m)
#pragma unroll
            for (int j = 0; j < 2; ++j) {
                const int col = (nt0 + j) * 16 + lrow;
#pragma unroll
                for (int rr = 0; rr < 8; ++rr) {       // C layout: M = rr + 8*lhi
                    float v = acc[m][j][rr] + bias1[j];
                    v = v > 0.0f ? v : 0.0f;
                    sH[(m * 16 + rr + 8 * lhi) * SH_STRIDE + col] = f2bf(v);
                }
            }
    }
    __syncthreads();

    // ---------------- stage 2: out = h @ W2 + b2 ---------------------------
    // 16 output tiles (4 M x 4 N); each wave does 1 M-subtile x 2 N-tiles.
    const int m2  = wave & 3;
    const int ntB = (wave >> 2) * 2;
    v8f acc2[2];
    acc2[0] = {}; acc2[1] = {};

#pragma unroll
    for (int ks = 0; ks < D_HID / 32; ++ks) {          // 8 K-steps of 32
        const int kc = ks * 32;
        Frag a;
        const unsigned short* ap = &sH[(m2 * 16 + lrow) * SH_STRIDE + kc + koff];
        a.u[0] = *(const v4u*)ap;
        a.u[1] = *(const v4u*)(ap + 16);
#pragma unroll
        for (int j = 0; j < 2; ++j) {
            Frag b;
            const unsigned short* bp =
                w2t + (size_t)((ntB + j) * 16 + lrow) * D_HID + kc + koff;
            b.u[0] = *(const v4u*)bp;
            b.u[1] = *(const v4u*)(bp + 16);
            acc2[j] = __builtin_amdgcn_wmma_f32_16x16x32_bf16(
                false, a.v, false, b.v, (short)0, acc2[j], false, false);
        }
    }

    // bias + f32 store (lanes 0..15 cover 16 consecutive floats of a row)
#pragma unroll
    for (int j = 0; j < 2; ++j) {
        const int col  = (ntB + j) * 16 + lrow;
        const float bv = b2[col];
#pragma unroll
        for (int rr = 0; rr < 8; ++rr) {
            const int e = e0 + m2 * 16 + rr + 8 * lhi;
            if (e < n_edges)
                out[(size_t)e * D_OUT + col] = acc2[j][rr] + bv;
        }
    }
}

// ---------------------------------------------------------------------------
extern "C" void kernel_launch(void* const* d_in, const int* in_sizes, int n_in,
                              void* d_out, int out_size, void* d_ws, size_t ws_size,
                              hipStream_t stream) {
    const float* edges    = (const float*)d_in[0];
    const float* nodes    = (const float*)d_in[1];
    const float* globals_ = (const float*)d_in[2];
    const int*   recv     = (const int*)d_in[3];
    const int*   send     = (const int*)d_in[4];
    const int*   egi      = (const int*)d_in[5];
    const float* W1       = (const float*)d_in[6];
    const float* b1       = (const float*)d_in[7];
    const float* W2       = (const float*)d_in[8];
    const float* b2       = (const float*)d_in[9];
    float* out = (float*)d_out;

    const int n_edges = in_sizes[0] / D_EDGE;

    // workspace: bf16 W1^T [256][224] then bf16 W2^T [64][256]
    unsigned short* w1t = (unsigned short*)d_ws;
    unsigned short* w2t = w1t + (size_t)D_HID * D_IN;

    const int prep_total = D_HID * D_IN + D_OUT * D_HID;   // 73728
    prep_weights<<<(prep_total + 255) / 256, 256, 0, stream>>>(W1, W2, w1t, w2t);

    const int n_tiles = (n_edges + TILE_M - 1) / TILE_M;   // 12500
    edge_mlp_kernel<<<n_tiles, 256, 0, stream>>>(edges, nodes, globals_,
                                                 recv, send, egi,
                                                 w1t, b1, w2t, b2,
                                                 out, n_edges);
}
